// SVDQuantLinear_87849261072827
// MI455X (gfx1250) — 16-bit and 8-bit matrix operand order repaired and checked by reading, compiled, not run on hardware
//
#include <hip/hip_runtime.h>
#include <hip/hip_bf16.h>
#include <cstdint>


typedef __attribute__((ext_vector_type(4)))  int       v4i;
typedef __attribute__((ext_vector_type(8)))  int       v8i;
typedef __attribute__((ext_vector_type(8)))  float     v8f;
typedef __attribute__((ext_vector_type(16))) _Float16  v16h;

#define IN_F   4096
#define OUT_F  4096
#define TOKENS 4096
#define RANK   32
#define GROUP  64
#define KGROUPS (IN_F / GROUP)

#if __has_builtin(__builtin_amdgcn_global_load_async_to_lds_b128) && \
    __has_builtin(__builtin_amdgcn_s_wait_asynccnt)
#define USE_ASYNC_LDS 1
#else
#define USE_ASYNC_LDS 0
#endif

#if USE_ASYNC_LDS
typedef __attribute__((address_space(1))) v4i* gv4i_p;
typedef __attribute__((address_space(3))) v4i* lv4i_p;

__device__ __forceinline__ void async_cp16(const void* g, void* l) {
  __builtin_amdgcn_global_load_async_to_lds_b128(
      (gv4i_p)(void*)g, (lv4i_p)l,  0,  0);
}
#endif

__global__ __launch_bounds__(256) void pack_w_kernel(const int* __restrict__ qw,
                                                     int* __restrict__ qpk) {
  size_t gid = (size_t)blockIdx.x * 256 + threadIdx.x;
  int4 c = reinterpret_cast<const int4*>(qw)[gid];
  qpk[gid] = (c.x & 0xFF) | ((c.y & 0xFF) << 8) | ((c.z & 0xFF) << 16) | ((c.w & 0xFF) << 24);
}

__global__ __launch_bounds__(256) void cvt_lup_kernel(const float* __restrict__ lu,
                                                      _Float16* __restrict__ lup_h) {
  int gid = blockIdx.x * 256 + threadIdx.x;
  lup_h[gid] = (_Float16)lu[gid];
}

__global__ __launch_bounds__(256) void quant_lora_kernel(const float* __restrict__ x,
                                                         const float* __restrict__ ld,
                                                         signed char* __restrict__ xq,
                                                         float* __restrict__ ascale,
                                                         _Float16* __restrict__ tmp_h) {
  __shared__ float sred[256];
  __shared__ float lacc[RANK];
  const int t   = blockIdx.x;
  const int tid = threadIdx.x;
  const int lane = tid & 31;

  const float4* xr4 = reinterpret_cast<const float4*>(x + (size_t)t * IN_F);
  float xv[16];
  float amax = 0.f;
#pragma unroll
  for (int c = 0; c < 4; ++c) {
    float4 v = xr4[c * 256 + tid];
    xv[c * 4 + 0] = v.x; xv[c * 4 + 1] = v.y; xv[c * 4 + 2] = v.z; xv[c * 4 + 3] = v.w;
    amax = fmaxf(amax, fmaxf(fmaxf(fabsf(v.x), fabsf(v.y)), fmaxf(fabsf(v.z), fabsf(v.w))));
  }
  sred[tid] = amax;
  __syncthreads();
#pragma unroll
  for (int s = 128; s > 0; s >>= 1) {
    if (tid < s) sred[tid] = fmaxf(sred[tid], sred[tid + s]);
    __syncthreads();
  }
  const float asc = fmaxf(sred[0], 1e-6f) / 7.0f;
  const float inv = 1.0f / asc;
  if (tid == 0) ascale[t] = asc;
  if (tid < RANK) lacc[tid] = 0.f;
  __syncthreads();

  int* xqi = reinterpret_cast<int*>(xq) + (size_t)t * (IN_F / 4);
#pragma unroll
  for (int c = 0; c < 4; ++c) {
    int p = 0;
#pragma unroll
    for (int j = 0; j < 4; ++j) {
      int q = (int)rintf(xv[c * 4 + j] * inv);
      q = q < -8 ? -8 : (q > 7 ? 7 : q);
      p |= (q & 0xFF) << (8 * j);
    }
    xqi[c * 256 + tid] = p;
  }

#pragma unroll
  for (int rg = 0; rg < 4; ++rg) {
    float p[8];
#pragma unroll
    for (int rr = 0; rr < 8; ++rr) {
      const int r = rg * 8 + rr;
      const float4* ldr4 = reinterpret_cast<const float4*>(ld + (size_t)r * IN_F);
      float acc = 0.f;
#pragma unroll
      for (int c = 0; c < 4; ++c) {
        float4 l4 = ldr4[c * 256 + tid];
        acc = fmaf(xv[c * 4 + 0], l4.x, acc);
        acc = fmaf(xv[c * 4 + 1], l4.y, acc);
        acc = fmaf(xv[c * 4 + 2], l4.z, acc);
        acc = fmaf(xv[c * 4 + 3], l4.w, acc);
      }
      p[rr] = acc;
    }
#pragma unroll
    for (int rr = 0; rr < 8; ++rr) {
#pragma unroll
      for (int o = 16; o > 0; o >>= 1) p[rr] += __shfl_xor(p[rr], o, 32);
      if (lane == 0) atomicAdd(&lacc[rg * 8 + rr], p[rr]);
    }
  }
  __syncthreads();
  if (tid < RANK) tmp_h[(size_t)t * RANK + tid] = (_Float16)lacc[tid];
}

__global__ __launch_bounds__(256) void svdq_gemm_kernel(
    const signed char* __restrict__ xq,
    const signed char* __restrict__ qpk,
    const float* __restrict__ wscale,
    const float* __restrict__ ascale,
    const _Float16* __restrict__ tmp_h,
    const _Float16* __restrict__ lup_h,
    const float* __restrict__ bias,
    float* __restrict__ out)
{
  __shared__ __align__(16) signed char sA[2][128 * 64];
  __shared__ __align__(16) signed char sB[2][128 * 64];
  __shared__ float sWs[2][128];
  __shared__ float sAs[128];
  __shared__ float sBias[128];
  __shared__ __align__(16) _Float16 sTmp[128 * RANK];
  __shared__ __align__(16) _Float16 sLup[128 * RANK];

  const int tid  = threadIdx.x;
  const int lane = tid & 31;
  const int wv   = tid >> 5;
  const int wm   = (wv & 1) * 64;
  const int wn   = (wv >> 1) * 32;
  const int l15  = lane & 15;
  const int lhi  = lane >> 4;

  const int T0 = blockIdx.y * 128;
  const int O0 = blockIdx.x * 128;

  auto stage_issue = [&](int g, int buf) {
    const int kbase = g * GROUP;
#pragma unroll
    for (int it = 0; it < 2; ++it) {
      const int lin = tid + it * 256;
      const int row = lin >> 2;
      const int ch  = lin & 3;
      const signed char* ga = xq  + (size_t)(T0 + row) * IN_F + kbase + ch * 16;
      const signed char* gb = qpk + (size_t)(O0 + row) * IN_F + kbase + ch * 16;
#if USE_ASYNC_LDS
      async_cp16(ga, &sA[buf][row * 64 + ch * 16]);
      async_cp16(gb, &sB[buf][row * 64 + ch * 16]);
#else
      *reinterpret_cast<int4*>(&sA[buf][row * 64 + ch * 16]) =
          *reinterpret_cast<const int4*>(ga);
      *reinterpret_cast<int4*>(&sB[buf][row * 64 + ch * 16]) =
          *reinterpret_cast<const int4*>(gb);
#endif
    }
    if (tid < 128) sWs[buf][tid] = wscale[(size_t)(O0 + tid) * KGROUPS + g];
  };

  float acc[4][2][8];
#pragma unroll
  for (int am = 0; am < 4; ++am)
#pragma unroll
    for (int bt = 0; bt < 2; ++bt)
#pragma unroll
      for (int e = 0; e < 8; ++e) acc[am][bt][e] = 0.f;

  stage_issue(0, 0);

  for (int g = 0; g < KGROUPS; ++g) {
    const int cur = g & 1;

    if (g + 1 < KGROUPS) {
      stage_issue(g + 1, cur ^ 1);
      if (g + 2 < KGROUPS) {
        __builtin_prefetch(xq  + (size_t)(T0 + (tid & 127)) * IN_F + (g + 2) * GROUP, 0, 1);
        __builtin_prefetch(qpk + (size_t)(O0 + (tid & 127)) * IN_F + (g + 2) * GROUP, 0, 1);
      }
    }
#if USE_ASYNC_LDS
    if (g + 1 < KGROUPS) {
      __builtin_amdgcn_s_wait_asynccnt(4);
    } else {
      __builtin_amdgcn_s_wait_asynccnt(0);
    }
#endif
    __syncthreads();

    v8i afr[4];
#pragma unroll
    for (int am = 0; am < 4; ++am) {
      const signed char* p = &sA[cur][(wm + am * 16 + l15) * 64];
      const int sh = lhi * 16;
      int2 d0 = *reinterpret_cast<const int2*>(p + 0  + sh);
      int2 d1 = *reinterpret_cast<const int2*>(p + 8  + sh);
      int2 d2 = *reinterpret_cast<const int2*>(p + 32 + sh);
      int2 d3 = *reinterpret_cast<const int2*>(p + 40 + sh);
      v8i a = {d0.x, d0.y, d1.x, d1.y, d2.x, d2.y, d3.x, d3.y};
      afr[am] = a;
    }
    v8i bfr[2];
    float wsc[2];
#pragma unroll
    for (int bt = 0; bt < 2; ++bt) {
      const signed char* p = &sB[cur][(wn + bt * 16 + l15) * 64];
      int4 q0 = *reinterpret_cast<const int4*>(p + lhi * 16);
      int4 q1 = *reinterpret_cast<const int4*>(p + 32 + lhi * 16);
      v8i b = {q0.x, q0.y, q0.z, q0.w, q1.x, q1.y, q1.z, q1.w};
      bfr[bt] = b;
      wsc[bt] = sWs[cur][wn + bt * 16 + l15];
    }

#pragma unroll
    for (int am = 0; am < 4; ++am) {
#pragma unroll
      for (int bt = 0; bt < 2; ++bt) {
        v8i z = {0, 0, 0, 0, 0, 0, 0, 0};
        v8i r = __builtin_amdgcn_wmma_i32_16x16x64_iu8(
             true, afr[am],  true, bfr[bt], z,
             false,  false);
#pragma unroll
        for (int e = 0; e < 8; ++e)
          acc[am][bt][e] = fmaf((float)r[e], wsc[bt], acc[am][bt][e]);
      }
    }
    __syncthreads();
  }

  if (tid < 128) {
    sAs[tid]   = ascale[T0 + tid];
    sBias[tid] = bias[O0 + tid];
  }
#pragma unroll
  for (int it = 0; it < 2; ++it) {
    const int lin = tid + it * 256;
    const int row = lin >> 2;
    const int ch  = lin & 3;
    *reinterpret_cast<int4*>(&sTmp[row * RANK + ch * 8]) =
        *(reinterpret_cast<const int4*>(tmp_h + (size_t)(T0 + row) * RANK) + ch);
    *reinterpret_cast<int4*>(&sLup[row * RANK + ch * 8]) =
        *(reinterpret_cast<const int4*>(lup_h + (size_t)(O0 + row) * RANK) + ch);
  }
  __syncthreads();

  v16h ah[4];
#pragma unroll
  for (int am = 0; am < 4; ++am) {
    const char* p = reinterpret_cast<const char*>(&sTmp[(wm + am * 16 + l15) * RANK]);
    int4 t0 = *reinterpret_cast<const int4*>(p + lhi * 16);
    int4 t1 = *reinterpret_cast<const int4*>(p + 32 + lhi * 16);
    v8i ti = {t0.x, t0.y, t0.z, t0.w, t1.x, t1.y, t1.z, t1.w};
    ah[am] = __builtin_bit_cast(v16h, ti);
  }
  v16h bh[2];
#pragma unroll
  for (int bt = 0; bt < 2; ++bt) {
    const char* p = reinterpret_cast<const char*>(&sLup[(wn + bt * 16 + l15) * RANK]);
    int4 b0 = *reinterpret_cast<const int4*>(p + lhi * 16);
    int4 b1 = *reinterpret_cast<const int4*>(p + 32 + lhi * 16);
    v8i bi = {b0.x, b0.y, b0.z, b0.w, b1.x, b1.y, b1.z, b1.w};
    bh[bt] = __builtin_bit_cast(v16h, bi);
  }

#pragma unroll
  for (int am = 0; am < 4; ++am) {
#pragma unroll
    for (int bt = 0; bt < 2; ++bt) {
      v8f c;
#pragma unroll
      for (int e = 0; e < 8; ++e)
        c[e] = acc[am][bt][e] * sAs[wm + am * 16 + e + lhi * 8];
      v8f d = __builtin_amdgcn_wmma_f32_16x16x32_f16(
           false, ah[am],  false, bh[bt],
           (short)0, c,  false,  false);
      const float bv = sBias[wn + bt * 16 + l15];
      const int og = O0 + wn + bt * 16 + l15;
      const int tg = T0 + wm + am * 16 + lhi * 8;
#pragma unroll
      for (int e = 0; e < 8; ++e)
        out[(size_t)(tg + e) * OUT_F + og] = d[e] + bv;
    }
  }
}

extern "C" void kernel_launch(void* const* d_in, const int* in_sizes, int n_in,
                              void* d_out, int out_size, void* d_ws, size_t ws_size,
                              hipStream_t stream) {
  const float* x     = (const float*)d_in[0];
  const int*   qw    = (const int*)d_in[1];
  const float* wsc   = (const float*)d_in[2];
  const float* ldown = (const float*)d_in[3];
  const float* lup   = (const float*)d_in[4];
  const float* bias  = (const float*)d_in[5];
  float* out = (float*)d_out;

  char* ws = (char*)d_ws;
  signed char* xq    = (signed char*)(ws);
  signed char* qpk   = (signed char*)(ws + (16u << 20));
  float*       asc   = (float*)(ws + (32u << 20));
  _Float16*    tmp_h = (_Float16*)(ws + (32u << 20) + (64u << 10));
  _Float16*    lup_h = (_Float16*)(ws + (32u << 20) + (512u << 10));

  (void)in_sizes; (void)n_in; (void)out_size; (void)ws_size;

  pack_w_kernel<<<dim3((OUT_F * (IN_F / 4)) / 256), dim3(256), 0, stream>>>(qw, (int*)qpk);
  cvt_lup_kernel<<<dim3((OUT_F * RANK) / 256), dim3(256), 0, stream>>>(lup, lup_h);
  quant_lora_kernel<<<dim3(TOKENS), dim3(256), 0, stream>>>(x, ldown, xq, asc, tmp_h);
  svdq_gemm_kernel<<<dim3(OUT_F / 128, TOKENS / 128), dim3(256), 0, stream>>>(
      xq, qpk, wsc, asc, tmp_h, lup_h, bias, out);
}
